// MFI_Block_20959440404534
// MI455X (gfx1250) — compile-verified, matrix operand order checked by reading, not run on hardware
//
#include <hip/hip_runtime.h>

// ---------------------------------------------------------------------------
// CDNA5 (gfx1250) transformer block: LN -> QKV proj -> mix -> flash-attn ->
// out proj (+res) -> LN -> MLP(GELU) (+res).  All GEMMs via
// v_wmma_f32_16x16x32_bf16 (bf16 in, f32 accumulate).  wave32.
// GEMM tiles staged with the Tensor Data Mover (tensor_load_to_lds),
// attention tiles with global_load_async_to_lds_b128 (ASYNCcnt).
// ---------------------------------------------------------------------------

typedef __attribute__((ext_vector_type(16))) __bf16   v16bf;
typedef __attribute__((ext_vector_type(8)))  float    v8f;
typedef __attribute__((ext_vector_type(4)))  unsigned u32x4;
typedef __attribute__((ext_vector_type(8)))  int      i32x8;
typedef __attribute__((ext_vector_type(4)))  int      i32x4;
typedef int gv4i __attribute__((vector_size(16)));    // matches builtin param

#define AS1 __attribute__((address_space(1)))
#define AS3 __attribute__((address_space(3)))

#if __has_builtin(__builtin_amdgcn_global_load_async_to_lds_b128) && \
    __has_builtin(__builtin_amdgcn_s_wait_asynccnt)
#define HAVE_ASYNC_LDS 1
#else
#define HAVE_ASYNC_LDS 0
#endif

#if __has_builtin(__builtin_amdgcn_tensor_load_to_lds) && \
    __has_builtin(__builtin_amdgcn_s_wait_tensorcnt)
#define HAVE_TDM 1
#else
#define HAVE_TDM 0
#endif

union FragAB {            // one 16x32 bf16 A/B operand fragment (8 VGPRs/lane)
    v16bf v;
    u32x4 u[2];
};

__device__ __forceinline__ unsigned short f2bf(float f) {
    union { float f; unsigned u; } c; c.f = f;
    unsigned u = c.u + 0x7fffu + ((c.u >> 16) & 1u);   // round-to-nearest-even
    return (unsigned short)(u >> 16);
}

__device__ __forceinline__ v8f wmma_bf16(v16bf a, v16bf b, v8f c) {
    // 8 args: (neg_a, A, neg_b, B, c_mod, C, reuse_a, reuse_b)
    return __builtin_amdgcn_wmma_f32_16x16x32_bf16(false, a, false, b,
                                                   (short)0, c, false, false);
}

// 16-byte global -> LDS stage (async on CDNA5, sync fallback elsewhere)
__device__ __forceinline__ void stage16(const unsigned short* g,
                                        unsigned short* l) {
#if HAVE_ASYNC_LDS
    __builtin_amdgcn_global_load_async_to_lds_b128(
        (AS1 gv4i*)g, (AS3 gv4i*)l, 0, 0);
#else
    *(u32x4*)l = *(const u32x4*)g;
#endif
}

__device__ __forceinline__ void async_wait() {
#if HAVE_ASYNC_LDS
    __builtin_amdgcn_s_wait_asynccnt(0);
#endif
}

#if HAVE_TDM
__device__ __forceinline__ unsigned lds_byte_off(const void* p) {
    return (unsigned)(size_t)(const AS3 void*)p;
}

// 2-D TDM tile load: tile_rows x tile_cols (bf16 elements) from a row-major
// tensor with row length/stride row_elems, packed contiguously into LDS.
__device__ __forceinline__ void tdm_load_2d(const unsigned short* g,
                                            unsigned lds_off, int row_elems,
                                            int tile_cols, int tile_rows) {
    unsigned long long ga = (unsigned long long)(size_t)g;
    u32x4 g0;
    g0.x = 1u;                                     // count=1, user descriptor
    g0.y = lds_off;                                // lds_addr (bytes)
    g0.z = (unsigned)ga;                           // global_addr[31:0]
    g0.w = (unsigned)(ga >> 32) | (2u << 30);      // global_addr[56:32], type=2
    i32x8 g1 = {0, 0, 0, 0, 0, 0, 0, 0};
    g1[0] = 1 << 16;                               // data_size = 2 bytes
    g1[1] = (int)(((unsigned)row_elems & 0xffffu) << 16);   // tensor_dim0 lo
    g1[2] = (int)((unsigned)row_elems >> 16);               // tensor_dim0 hi
    g1[3] = (int)(((1u << 20) >> 16) |                      // tensor_dim1 hi
                  ((unsigned)tile_cols << 16));             // tile_dim0
    g1[4] = tile_rows & 0xffff;                             // tile_dim1
    g1[5] = row_elems;                                      // dim0_stride lo32
    i32x4 z4 = {0, 0, 0, 0};
    i32x8 z8 = {0, 0, 0, 0, 0, 0, 0, 0};
    // amdgpu-toolchain 6-arg form: (g0, g1, g2, g3, g_extra, cpol)
    __builtin_amdgcn_tensor_load_to_lds(g0, g1, z4, z4, z8, 0);
}
#endif

// ---------------------------------------------------------------------------
// fp32 -> bf16 bulk convert (weights)
// ---------------------------------------------------------------------------
__global__ void cvt_bf16_kernel(const float* __restrict__ in,
                                unsigned short* __restrict__ out, unsigned n) {
    unsigned i = blockIdx.x * 256u + threadIdx.x;
    if (i < n) out[i] = f2bf(in[i]);
}

// ---------------------------------------------------------------------------
// LayerNorm over C=1024, one block per row, bf16 output
// ---------------------------------------------------------------------------
__global__ __launch_bounds__(256) void ln_kernel(const float* __restrict__ x,
                                                 const float* __restrict__ gw,
                                                 const float* __restrict__ gb,
                                                 unsigned short* __restrict__ out) {
    __shared__ float red[16];
    const int row  = blockIdx.x;
    const int tid  = threadIdx.x, lane = tid & 31, wv = tid >> 5;
    const float* xr = x + (size_t)row * 1024;
    float4 v = *(const float4*)(xr + tid * 4);

    float s = v.x + v.y + v.z + v.w;
#pragma unroll
    for (int m = 16; m >= 1; m >>= 1) s += __shfl_xor(s, m, 32);
    if (lane == 0) red[wv] = s;
    __syncthreads();
    float tot = 0.f;
#pragma unroll
    for (int i = 0; i < 8; ++i) tot += red[i];
    const float mu = tot * (1.f / 1024.f);

    float dx = v.x - mu, dy = v.y - mu, dz = v.z - mu, dw = v.w - mu;
    float q = dx * dx + dy * dy + dz * dz + dw * dw;
#pragma unroll
    for (int m = 16; m >= 1; m >>= 1) q += __shfl_xor(q, m, 32);
    __syncthreads();
    if (lane == 0) red[wv] = q;
    __syncthreads();
    float vt = 0.f;
#pragma unroll
    for (int i = 0; i < 8; ++i) vt += red[i];
    const float rs = rsqrtf(vt * (1.f / 1024.f) + 1e-5f);

    const int c0 = tid * 4;
    float4 w4 = *(const float4*)(gw + c0);
    float4 b4 = *(const float4*)(gb + c0);
    unsigned o0 = f2bf(dx * rs * w4.x + b4.x);
    unsigned o1 = f2bf(dy * rs * w4.y + b4.y);
    unsigned o2 = f2bf(dz * rs * w4.z + b4.z);
    unsigned o3 = f2bf(dw * rs * w4.w + b4.w);
    uint2 pk; pk.x = o0 | (o1 << 16); pk.y = o2 | (o3 << 16);
    *(uint2*)(out + (size_t)row * 1024 + c0) = pk;
}

// ---------------------------------------------------------------------------
// Generic WMMA GEMM: Y[M,N] = A[M,K](bf16) @ W[N,K](bf16)^T + bias
//   mode 0: outF = acc + bias (+ residual)     (fp32)
//   mode 1: outBF = bf16(gelu(acc + bias))     (MLP hidden)
// 256 thr = 8 waves, tile 128x128, K-step 32, double-buffered LDS staged by
// the Tensor Data Mover (wave 0 issues descriptors, overlapped with WMMA).
// ---------------------------------------------------------------------------
__global__ __launch_bounds__(256) void gemm_bf16_wmma(
        const unsigned short* __restrict__ A,
        const unsigned short* __restrict__ W,
        const float* __restrict__ bias,
        const float* __restrict__ residual,
        float* __restrict__ outF,
        unsigned short* __restrict__ outBF,
        int M, int N, int K, int mode) {
    __shared__ __align__(16) unsigned short lds_a[2][128 * 32];
    __shared__ __align__(16) unsigned short lds_b[2][128 * 32];

    const int tid  = threadIdx.x;
    const int lane = tid & 31, wv = tid >> 5;
    const int r16  = lane & 15, hi = lane >> 4;
    const int koff = hi * 8;
    const int m0 = blockIdx.y * 128, n0 = blockIdx.x * 128;
    const int mwave = (wv & 3) * 32;       // 4 waves along M
    const int nwave = (wv >> 2) * 64;      // 2 waves along N

    auto stage_tiles = [&](int buf, int k0) {
#if HAVE_TDM
        if (wv == 0) {   // wave-uniform; TDM ignores EXEC, one issue per wave
            tdm_load_2d(A + (size_t)m0 * K + k0, lds_byte_off(lds_a[buf]),
                        K, 32, 128);
            tdm_load_2d(W + (size_t)n0 * K + k0, lds_byte_off(lds_b[buf]),
                        K, 32, 128);
        }
#else
#pragma unroll
        for (int i = 0; i < 2; ++i) {
            int cid = tid + i * 256;
            int r = cid >> 2, cc = cid & 3;
            *(u32x4*)&lds_a[buf][r * 32 + cc * 8] =
                *(const u32x4*)(A + (size_t)(m0 + r) * K + k0 + cc * 8);
            *(u32x4*)&lds_b[buf][r * 32 + cc * 8] =
                *(const u32x4*)(W + (size_t)(n0 + r) * K + k0 + cc * 8);
        }
#endif
    };
    auto stage_commit = [&]() {
#if HAVE_TDM
        if (wv == 0) __builtin_amdgcn_s_wait_tensorcnt(0);
#endif
        __syncthreads();
    };

    const v8f zacc = {0.f, 0.f, 0.f, 0.f, 0.f, 0.f, 0.f, 0.f};
    v8f acc[2][4];
#pragma unroll
    for (int mi = 0; mi < 2; ++mi)
#pragma unroll
        for (int ni = 0; ni < 4; ++ni) acc[mi][ni] = zacc;

    stage_tiles(0, 0);
    stage_commit();

    const int nk = K >> 5;
    for (int it = 0; it < nk; ++it) {
        const int cur = it & 1;
        if (it + 1 < nk) {
            stage_tiles(cur ^ 1, (it + 1) << 5);        // overlap with WMMA
            __builtin_prefetch(A + (size_t)(m0 + (tid >> 1)) * K + ((it + 1) << 5), 0, 1);
            __builtin_prefetch(W + (size_t)(n0 + (tid >> 1)) * K + ((it + 1) << 5), 0, 1);
        }

        FragAB af[2], bf[4];
#pragma unroll
        for (int mi = 0; mi < 2; ++mi) {
            const unsigned short* p =
                &lds_a[cur][(mwave + mi * 16 + r16) * 32 + koff];
            af[mi].u[0] = *(const u32x4*)p;
            af[mi].u[1] = *(const u32x4*)(p + 16);
        }
#pragma unroll
        for (int ni = 0; ni < 4; ++ni) {
            const unsigned short* p =
                &lds_b[cur][(nwave + ni * 16 + r16) * 32 + koff];
            bf[ni].u[0] = *(const u32x4*)p;
            bf[ni].u[1] = *(const u32x4*)(p + 16);
        }
#pragma unroll
        for (int mi = 0; mi < 2; ++mi)
#pragma unroll
            for (int ni = 0; ni < 4; ++ni)
                acc[mi][ni] = wmma_bf16(af[mi].v, bf[ni].v, acc[mi][ni]);

        stage_commit();   // next buffer ready + this buffer's readers done
    }

    // epilogue: C/D layout -> lane holds col n=lane%16, rows in VGPRs (+8 hi)
#pragma unroll
    for (int mi = 0; mi < 2; ++mi) {
#pragma unroll
        for (int ni = 0; ni < 4; ++ni) {
            const int col = n0 + nwave + ni * 16 + r16;
            const float bs = bias ? bias[col] : 0.f;
#pragma unroll
            for (int v = 0; v < 8; ++v) {
                const int row = m0 + mwave + mi * 16 + v + hi * 8;
                const size_t idx = (size_t)row * N + col;
                float val = acc[mi][ni][v] + bs;
                if (mode == 0) {
                    if (residual) val += residual[idx];
                    outF[idx] = val;
                } else {
                    const float g = 0.5f * val * (1.f + erff(val * 0.70710678118f));
                    outBF[idx] = f2bf(g);
                }
            }
        }
    }
}

// ---------------------------------------------------------------------------
// Mix: Qm=.7*Qproj+.3*Qin (etc), write fp32 outputs + bf16 Q/K head-major and
// V transposed [b,h,hs,n] for the P@V WMMA B operand.
// ---------------------------------------------------------------------------
__global__ void mix_kernel(const float* __restrict__ qk,
                           const float* __restrict__ vproj,
                           const float* __restrict__ Qin,
                           const float* __restrict__ Kin,
                           const float* __restrict__ Vin,
                           float* __restrict__ Qout, float* __restrict__ Kout,
                           float* __restrict__ Vout,
                           unsigned short* __restrict__ Qb,
                           unsigned short* __restrict__ Kb,
                           unsigned short* __restrict__ Vt) {
    size_t idx = (size_t)blockIdx.x * 256 + threadIdx.x;
    if (idx >= (size_t)4 * 16 * 1024 * 64) return;
    const int hs = idx & 63;
    size_t t = idx >> 6;
    const int n = t & 1023; t >>= 10;
    const int h = t & 15;
    const int b = (int)(t >> 4);
    const size_t rowbn = (size_t)b * 1024 + n;
    const float qp = qk[rowbn * 2048 + (size_t)(h * 64 + hs) * 2 + 0];
    const float kp = qk[rowbn * 2048 + (size_t)(h * 64 + hs) * 2 + 1];
    const float vp = vproj[rowbn * 1024 + h * 64 + hs];
    const float qm = 0.7f * qp + 0.3f * Qin[idx];
    const float km = 0.7f * kp + 0.3f * Kin[idx];
    const float vm = 0.7f * vp + 0.3f * Vin[idx];
    Qout[idx] = qm; Kout[idx] = km; Vout[idx] = vm;
    Qb[idx] = f2bf(qm); Kb[idx] = f2bf(km);
    Vt[((((size_t)b * 16 + h) * 64 + hs) * 1024) + n] = f2bf(vm);
}

// ---------------------------------------------------------------------------
// Fused flash-attention.  Grid: B*H*(N/64) blocks, 128 thr = 4 waves.
// K/V^T tiles staged via async-to-LDS (ASYNCcnt); S via 2x2 WMMAs (K fed as
// B operand), online softmax, P through LDS (acc layout -> A layout),
// O += P@V via 4 WMMAs (V^T tile as B operand).
// ---------------------------------------------------------------------------
__global__ __launch_bounds__(128) void attn_kernel(
        const unsigned short* __restrict__ Qb,
        const unsigned short* __restrict__ Kb,
        const unsigned short* __restrict__ Vt,
        unsigned short* __restrict__ Ob) {
    __shared__ __align__(16) unsigned short kt[32 * 64];      // [key][hs]
    __shared__ __align__(16) unsigned short vtile[64 * 32];   // [hs][key]
    __shared__ __align__(16) unsigned short pst[4][16 * 32];  // per-wave P

    const int NSEQ = 1024, HS = 64;
    const int bh = blockIdx.x >> 4, qblk = blockIdx.x & 15;
    const int b = bh >> 4, h = bh & 15;
    const int tid = threadIdx.x, lane = tid & 31, wv = tid >> 5;
    const int r16 = lane & 15, hi = lane >> 4, koff = hi * 8;
    const int q0 = qblk * 64 + wv * 16;

    const unsigned short* qbase = Qb + (size_t)bh * NSEQ * HS;
    const unsigned short* kbase = Kb + (size_t)bh * NSEQ * HS;
    const unsigned short* vbase = Vt + (size_t)bh * HS * NSEQ;

    FragAB qf[2];
#pragma unroll
    for (int f = 0; f < 2; ++f) {
        const unsigned short* p = qbase + (size_t)(q0 + r16) * HS + f * 32 + koff;
        qf[f].u[0] = *(const u32x4*)p;
        qf[f].u[1] = *(const u32x4*)(p + 16);
    }

    const v8f zacc = {0.f, 0.f, 0.f, 0.f, 0.f, 0.f, 0.f, 0.f};
    v8f O[4]; float m_i[8], l_i[8];
#pragma unroll
    for (int c = 0; c < 4; ++c) O[c] = zacc;
#pragma unroll
    for (int v = 0; v < 8; ++v) { m_i[v] = -1e30f; l_i[v] = 0.f; }

    for (int j0 = 0; j0 < NSEQ; j0 += 32) {
        __syncthreads();   // previous tile fully consumed
        // K tile 32x64 and V^T tile 64x32: 256 x 16B chunks each, 2/thread
#pragma unroll
        for (int i = 0; i < 2; ++i) {
            int cid = tid + i * 128;
            int r = cid >> 3, cc = cid & 7;
            stage16(kbase + (size_t)(j0 + r) * HS + cc * 8,
                    &kt[r * 64 + cc * 8]);
            int r2 = cid >> 2, cc2 = cid & 3;
            stage16(vbase + (size_t)r2 * NSEQ + j0 + cc2 * 8,
                    &vtile[r2 * 32 + cc2 * 8]);
        }
        async_wait();
        __syncthreads();

        v8f S[2];
#pragma unroll
        for (int kg = 0; kg < 2; ++kg) {
            v8f s = zacc;
#pragma unroll
            for (int f = 0; f < 2; ++f) {
                FragAB kf;
                const unsigned short* p = &kt[(kg * 16 + r16) * 64 + f * 32 + koff];
                kf.u[0] = *(const u32x4*)p;
                kf.u[1] = *(const u32x4*)(p + 16);
                s = wmma_bf16(qf[f].v, kf.v, s);
            }
            S[kg] = s;
        }

        const float scale = 0.125f;   // 1/sqrt(64)
        float alpha[8];
#pragma unroll
        for (int v = 0; v < 8; ++v) {
            const float s0 = S[0][v] * scale, s1 = S[1][v] * scale;
            float rm = fmaxf(s0, s1);
#pragma unroll
            for (int m = 8; m >= 1; m >>= 1) rm = fmaxf(rm, __shfl_xor(rm, m, 32));
            const float mn = fmaxf(m_i[v], rm);
            const float a  = __expf(m_i[v] - mn);
            const float p0 = __expf(s0 - mn), p1 = __expf(s1 - mn);
            float rs = p0 + p1;
#pragma unroll
            for (int m = 8; m >= 1; m >>= 1) rs += __shfl_xor(rs, m, 32);
            l_i[v] = l_i[v] * a + rs;
            m_i[v] = mn; alpha[v] = a;
            unsigned short* pw = &pst[wv][(v + hi * 8) * 32];
            pw[r16]      = f2bf(p0);
            pw[16 + r16] = f2bf(p1);
        }
#pragma unroll
        for (int c = 0; c < 4; ++c)
#pragma unroll
            for (int v = 0; v < 8; ++v) O[c][v] *= alpha[v];
        __syncthreads();

        FragAB pf;
        const unsigned short* pp = &pst[wv][r16 * 32 + koff];
        pf.u[0] = *(const u32x4*)pp;
        pf.u[1] = *(const u32x4*)(pp + 16);
#pragma unroll
        for (int c = 0; c < 4; ++c) {
            FragAB vf;
            const unsigned short* vp = &vtile[(c * 16 + r16) * 32 + koff];
            vf.u[0] = *(const u32x4*)vp;
            vf.u[1] = *(const u32x4*)(vp + 16);
            O[c] = wmma_bf16(pf.v, vf.v, O[c]);
        }
    }

    // write o in (B,N,C) bf16 for the out-projection GEMM
#pragma unroll
    for (int v = 0; v < 8; ++v) {
        const float inv = 1.f / l_i[v];
        const int q = q0 + v + hi * 8;
        unsigned short* orow = Ob + ((size_t)b * NSEQ + q) * 1024 + h * 64;
#pragma unroll
        for (int c = 0; c < 4; ++c) orow[c * 16 + r16] = f2bf(O[c][v] * inv);
    }
}

// ---------------------------------------------------------------------------
// Host orchestration
// ---------------------------------------------------------------------------
extern "C" void kernel_launch(void* const* d_in, const int* in_sizes, int n_in,
                              void* d_out, int out_size, void* d_ws, size_t ws_size,
                              hipStream_t stream) {
    (void)in_sizes; (void)n_in; (void)out_size; (void)ws_size;
    const float* x      = (const float*)d_in[0];
    const float* Qin    = (const float*)d_in[1];
    const float* Kin    = (const float*)d_in[2];
    const float* Vin    = (const float*)d_in[3];
    const float* ln_w   = (const float*)d_in[4];
    const float* ln_b   = (const float*)d_in[5];
    const float* qk_w   = (const float*)d_in[6];
    const float* qk_b   = (const float*)d_in[7];
    const float* v_w    = (const float*)d_in[8];
    const float* v_b    = (const float*)d_in[9];
    const float* out_w  = (const float*)d_in[10];
    const float* out_b  = (const float*)d_in[11];
    const float* mlp_w1 = (const float*)d_in[12];
    const float* mlp_b1 = (const float*)d_in[13];
    const float* mlp_w2 = (const float*)d_in[14];
    const float* mlp_b2 = (const float*)d_in[15];

    const size_t M4 = (size_t)4 * 1024 * 1024;   // 4M elements
    float* y    = (float*)d_out;
    float* Qout = y + M4;
    float* Kout = y + 2 * M4;
    float* Vout = y + 3 * M4;

    char* p = (char*)d_ws;
    auto carve = [&](size_t bytes) {
        void* r = (void*)p;
        p += (bytes + 255) & ~(size_t)255;
        return r;
    };
    unsigned short* xn   = (unsigned short*)carve(M4 * 2);          // LN1 bf16
    unsigned short* wqk  = (unsigned short*)carve(2 * M4);          // 2M bf16
    unsigned short* wvp  = (unsigned short*)carve(M4 / 2);          // 1M bf16
    unsigned short* wout = (unsigned short*)carve(M4 / 2);          // 1M bf16
    unsigned short* wm1  = (unsigned short*)carve(M4 * 2);          // 4M bf16
    unsigned short* wm2  = (unsigned short*)carve(M4 * 2);          // 4M bf16
    float*          qko  = (float*)carve(2 * M4 * 4);               // 8M f32
    float*          vo   = (float*)carve(M4 * 4);                   // 4M f32
    unsigned short* Qb   = (unsigned short*)carve(M4 * 2);
    unsigned short* Kb   = (unsigned short*)carve(M4 * 2);
    unsigned short* Vt   = (unsigned short*)carve(M4 * 2);
    unsigned short* ob   = (unsigned short*)carve(M4 * 2);
    float*          xmid = (float*)carve(M4 * 4);
    unsigned short* xn2  = (unsigned short*)carve(M4 * 2);
    unsigned short* hb   = (unsigned short*)carve((size_t)16 * 1024 * 1024 * 2);

    auto cvt = [&](const float* src, unsigned short* dst, unsigned n) {
        cvt_bf16_kernel<<<(n + 255) / 256, 256, 0, stream>>>(src, dst, n);
    };
    cvt(qk_w,   wqk, 2 * 1024 * 1024);
    cvt(v_w,    wvp, 1024 * 1024);
    cvt(out_w,  wout, 1024 * 1024);
    cvt(mlp_w1, wm1, 4 * 1024 * 1024);
    cvt(mlp_w2, wm2, 4 * 1024 * 1024);

    ln_kernel<<<4096, 256, 0, stream>>>(x, ln_w, ln_b, xn);

    // qk projection: 4096x2048, K=1024
    gemm_bf16_wmma<<<dim3(2048 / 128, 4096 / 128), 256, 0, stream>>>(
        xn, wqk, qk_b, nullptr, qko, nullptr, 4096, 2048, 1024, 0);
    // v projection: 4096x1024, K=1024
    gemm_bf16_wmma<<<dim3(1024 / 128, 4096 / 128), 256, 0, stream>>>(
        xn, wvp, v_b, nullptr, vo, nullptr, 4096, 1024, 1024, 0);

    mix_kernel<<<(unsigned)(M4 / 256), 256, 0, stream>>>(
        qko, vo, Qin, Kin, Vin, Qout, Kout, Vout, Qb, Kb, Vt);

    attn_kernel<<<4 * 16 * 16, 128, 0, stream>>>(Qb, Kb, Vt, ob);

    // out projection + residual(x): 4096x1024, K=1024
    gemm_bf16_wmma<<<dim3(1024 / 128, 4096 / 128), 256, 0, stream>>>(
        ob, wout, out_b, x, xmid, nullptr, 4096, 1024, 1024, 0);

    ln_kernel<<<4096, 256, 0, stream>>>(xmid, ln_w, ln_b, xn2);

    // MLP up + exact GELU -> bf16: 4096x4096, K=1024
    gemm_bf16_wmma<<<dim3(4096 / 128, 4096 / 128), 256, 0, stream>>>(
        xn2, wm1, mlp_b1, nullptr, nullptr, hb, 4096, 4096, 1024, 1);
    // MLP down + residual(xmid) -> y: 4096x1024, K=4096
    gemm_bf16_wmma<<<dim3(1024 / 128, 4096 / 128), 256, 0, stream>>>(
        hb, wm2, mlp_b2, xmid, y, nullptr, 4096, 1024, 4096, 0);
}
